// NeuronDbrxBlock_48954037240377
// MI455X (gfx1250) — matrix operand order repaired and checked by reading, not compiled, not run on hardware
//
#include <hip/hip_runtime.h>
#include <math.h>

typedef __bf16 bf16;
typedef __attribute__((ext_vector_type(16))) __bf16 v16bf;
typedef __attribute__((ext_vector_type(8)))  float  v8f;
typedef int v4i_ __attribute__((vector_size(4 * sizeof(int))));

constexpr int kBATCH = 2;
constexpr int kS   = 2048;
constexpr int kD   = 2048;
constexpr int kH   = 16;
constexpr int kHKV = 4;
constexpr int kHD  = 128;
constexpr int kE   = 8;
constexpr int kF   = 2048;
constexpr int kT   = kBATCH * kS;   // 4096 tokens

// ---------- small helpers ----------
__device__ __forceinline__ bf16 f2bf(float f) {
  unsigned u = __builtin_bit_cast(unsigned, f);
  unsigned r = u + 0x7FFFu + ((u >> 16) & 1u);           // round-to-nearest-even
  unsigned short h = (unsigned short)(r >> 16);
  return __builtin_bit_cast(bf16, h);
}
__device__ __forceinline__ float b2f(bf16 x) {
  unsigned short h = __builtin_bit_cast(unsigned short, x);
  unsigned u = ((unsigned)h) << 16;
  return __builtin_bit_cast(float, u);
}

__device__ __forceinline__ v8f wmma_bf16(v16bf a, v16bf b, v8f c) {
  // D = A(16x32 bf16) x B(32x16 bf16) + C(16x16 f32)
  return __builtin_amdgcn_wmma_f32_16x16x32_bf16(false, a, false, b, (short)0, c,
                                                 false, false);
}

// ---------- async global->LDS 16-byte copy (CDNA5 ASYNCcnt path) ----------
#if __has_builtin(__builtin_amdgcn_global_load_async_to_lds_b128)
#define ASYNC_LDS_COPY 1
#endif

__device__ __forceinline__ void lds_copy16(bf16* dst, const bf16* src) {
#ifdef ASYNC_LDS_COPY
  __builtin_amdgcn_global_load_async_to_lds_b128(
      (__attribute__((address_space(1))) v4i_*)(unsigned long long)(uintptr_t)src,
      (__attribute__((address_space(3))) v4i_*)dst, 0, 0);
#else
  *(v16bf*)dst = *(const v16bf*)src;
#endif
}

__device__ __forceinline__ void async_fence() {
#ifdef ASYNC_LDS_COPY
#if __has_builtin(__builtin_amdgcn_s_wait_asynccnt)
  __builtin_amdgcn_s_wait_asynccnt(0);
#else
  asm volatile("s_wait_asynccnt 0x0" ::: "memory");
#endif
#endif
}

// A-matrix fragment, K=32, from a contiguous 32-element K-row.
// Layout (matches frag_b32): half-lane 0 -> K 0..15 ; half-lane 1 -> K 16..31
__device__ __forceinline__ v16bf frag_a32(const bf16* rowp) {
  unsigned hl = (threadIdx.x >> 4) & 1u;
  const bf16* p0 = rowp + hl * 16;
  v16bf f;
#pragma unroll
  for (int i = 0; i < 16; ++i) f[i] = p0[i];
  return f;
}

// B-matrix fragment, K=32, from a contiguous 32-element K-row (B stored N-major).
// Contiguous k order (same as frag_a32): half-lane 0 -> K 0..15 ; half-lane 1 -> K 16..31
__device__ __forceinline__ v16bf frag_b32(const bf16* rowp) {
  unsigned hl = (threadIdx.x >> 4) & 1u;
  const bf16* p = rowp + hl * 16;
  v16bf f;
#pragma unroll
  for (int i = 0; i < 16; ++i) f[i] = p[i];
  return f;
}

// B-matrix fragment, K=32, strided (K-major source, e.g. V tile rows)
__device__ __forceinline__ v16bf frag_b32_strided(const bf16* base, int stride) {
  unsigned hl = (threadIdx.x >> 4) & 1u;
  v16bf f;
#pragma unroll
  for (int i = 0; i < 16; ++i) f[i] = base[(hl * 16 + i) * stride];
  return f;
}

// ---------- LayerNorm (no bias) -> bf16 ----------
__global__ __launch_bounds__(256) void layernorm_kernel(
    const float* __restrict__ X, const float* __restrict__ W, bf16* __restrict__ Y) {
  int row = blockIdx.x;
  const float* x = X + (size_t)row * kD;
  float s = 0.f, ss = 0.f;
  for (int d = threadIdx.x; d < kD; d += 256) { float v = x[d]; s += v; ss += v * v; }
#pragma unroll
  for (int m = 1; m < 32; m <<= 1) { s += __shfl_xor(s, m, 32); ss += __shfl_xor(ss, m, 32); }
  __shared__ float rs[8], rss[8];
  int w = threadIdx.x >> 5, lane = threadIdx.x & 31;
  if (lane == 0) { rs[w] = s; rss[w] = ss; }
  __syncthreads();
  s = 0.f; ss = 0.f;
#pragma unroll
  for (int i = 0; i < 8; ++i) { s += rs[i]; ss += rss[i]; }
  float mu = s / (float)kD;
  float var = ss / (float)kD - mu * mu;
  float inv = rsqrtf(var + 1e-5f);
  bf16* y = Y + (size_t)row * kD;
  for (int d = threadIdx.x; d < kD; d += 256)
    y[d] = f2bf((x[d] - mu) * inv * W[d]);
}

// ---------- generic bf16-WMMA GEMM: C[M,N] = A[M,K](bf16) * B[K,N](f32->bf16) ----------
// Block tile: 128(M) x 128(N) x 32(K); wave tile 16x128 (8 WMMA accumulators).
template <bool CLIP_OUT, bool ADD_RES>
__global__ __launch_bounds__(256) void gemm_kernel(
    const bf16* __restrict__ A, const float* __restrict__ Bw,
    const float* __restrict__ Res, float* __restrict__ C, int M, int N, int K) {
  __shared__ __align__(32) bf16 sA[128 * 32];
  __shared__ __align__(32) bf16 sB[128 * 32];   // N-major: sB[n*32 + k]
  int mBase = blockIdx.y * 128;
  int nBase = blockIdx.x * 128;
  int w = threadIdx.x >> 5, lane = threadIdx.x & 31;
  int ln = lane & 15, hl = lane >> 4;
  v8f zero = {};
  v8f acc[8];
#pragma unroll
  for (int i = 0; i < 8; ++i) acc[i] = zero;

  for (int k0 = 0; k0 < K; k0 += 32) {
    {  // A tile 128x32 (async DMA to LDS when available)
      int row = threadIdx.x >> 1;
      int ch = (threadIdx.x & 1) * 16;
      int gm = mBase + row;
      bf16* dst = &sA[row * 32 + ch];
      if (gm < M) {
        lds_copy16(dst, A + (size_t)gm * K + k0 + ch);
        if (k0 + 32 < K)
          __builtin_prefetch(A + (size_t)gm * K + k0 + 32 + ch, 0, 0);
      } else {
        v16bf zz = {};
        *(v16bf*)dst = zz;
      }
    }
    {  // B tile 32x128, convert f32->bf16 into N-major LDS
      int k = threadIdx.x >> 3;             // 0..31
      int n = (threadIdx.x & 7) * 16;       // 0..112
      const float* src = Bw + (size_t)(k0 + k) * N + nBase + n;
#pragma unroll
      for (int i = 0; i < 16; ++i) sB[(n + i) * 32 + k] = f2bf(src[i]);
    }
    async_fence();
    __syncthreads();
    v16bf af = frag_a32(&sA[(w * 16 + ln) * 32]);
#pragma unroll
    for (int nS = 0; nS < 8; ++nS) {
      v16bf bfg = frag_b32(&sB[(nS * 16 + ln) * 32]);
      acc[nS] = wmma_bf16(af, bfg, acc[nS]);
    }
    __syncthreads();
  }
#pragma unroll
  for (int nS = 0; nS < 8; ++nS)
#pragma unroll
    for (int r = 0; r < 8; ++r) {
      int m = w * 16 + r + 8 * hl;
      int gm = mBase + m, gn = nBase + nS * 16 + ln;
      if (gm < M) {
        float v = acc[nS][r];
        if (CLIP_OUT) v = fminf(8.0f, fmaxf(-8.0f, v));
        if (ADD_RES) v += Res[(size_t)gm * N + gn];
        C[(size_t)gm * N + gn] = v;
      }
    }
}

// ---------- RoPE (+ implicit bf16 convert); X,Y: [T, heads, 128] ----------
__global__ void rope_kernel(const float* __restrict__ X, bf16* __restrict__ Y, int heads) {
  size_t idx = (size_t)blockIdx.x * blockDim.x + threadIdx.x;
  size_t total = (size_t)kT * heads * 64;
  if (idx >= total) return;
  int j = (int)(idx & 63);
  size_t th = idx >> 6;
  int hh = (int)(th % heads);
  size_t t = th / heads;
  int pos = (int)(t & (kS - 1));
  // inv_freq = theta^(-2j/128) = exp(-j * ln(500000)/64)
  float fr = (float)pos * __expf(-(float)j * 0.20503692777194262f);
  float sn, cs;
  __sincosf(fr, &sn, &cs);
  size_t base = (t * heads + hh) * (size_t)kHD + j;
  float x1 = X[base], x2 = X[base + 64];
  Y[base]      = f2bf(x1 * cs - x2 * sn);
  Y[base + 64] = f2bf(x2 * cs + x1 * sn);
}

__global__ void f32_to_bf16_kernel(const float* __restrict__ X, bf16* __restrict__ Y, size_t n) {
  size_t i = (size_t)blockIdx.x * blockDim.x + threadIdx.x;
  if (i < n) Y[i] = f2bf(X[i]);
}

// ---------- flash attention (causal, GQA), all-WMMA ----------
__global__ __launch_bounds__(256) void flash_attn_kernel(
    const bf16* __restrict__ Qb, const bf16* __restrict__ Kb,
    const bf16* __restrict__ Vb, bf16* __restrict__ Ob) {
  __shared__ __align__(32) bf16 sK[32 * 128];
  __shared__ __align__(32) bf16 sV[32 * 128];
  __shared__ __align__(32) bf16 sP[8 * 16 * 32];

  const int w = threadIdx.x >> 5;
  const int lane = threadIdx.x & 31;
  const int hl = lane >> 4;
  const int ln = lane & 15;
  const int b = blockIdx.y >> 4;   // H = 16
  const int h = blockIdx.y & 15;
  const int kvh = h >> 2;          // rep = H/HKV = 4
  const int qtile = blockIdx.x;
  const int qbase = qtile * 128 + w * 16;

  // Q fragments (held in registers for the whole pass)
  v16bf qf[4];
  {
    const bf16* qp = Qb + ((size_t)b * kS + qbase + ln) * (kH * kHD) + h * kHD;
#pragma unroll
    for (int kf = 0; kf < 4; ++kf) qf[kf] = frag_a32(qp + kf * 32);
  }

  v8f zero = {};
  v8f o[8];
#pragma unroll
  for (int i = 0; i < 8; ++i) o[i] = zero;
  float mrow[8], lrow[8];
#pragma unroll
  for (int r = 0; r < 8; ++r) { mrow[r] = -3.0e38f; lrow[r] = 0.f; }

  const int ktiles = (qtile + 1) * 4;      // 32 keys per tile, causal
  for (int kt = 0; kt < ktiles; ++kt) {
    {  // cooperative K/V tile load (32 x 128 bf16 each), async DMA when available
      int row = threadIdx.x >> 3;
      int col = (threadIdx.x & 7) * 16;
      size_t krow = (size_t)b * kS + kt * 32 + row;
      lds_copy16(&sK[row * 128 + col], Kb + (krow * kHKV + kvh) * kHD + col);
      lds_copy16(&sV[row * 128 + col], Vb + (krow * kHKV + kvh) * kHD + col);
    }
    async_fence();
    __syncthreads();

    // S = Q K^T  (two 16-key sub-tiles)
    v8f st[2];
#pragma unroll
    for (int nS = 0; nS < 2; ++nS) {
      v8f s = zero;
#pragma unroll
      for (int kf = 0; kf < 4; ++kf) {
        v16bf bfg = frag_b32(&sK[(nS * 16 + ln) * 128 + kf * 32]);
        s = wmma_bf16(qf[kf], bfg, s);
      }
      st[nS] = s;
    }

    // scale + causal mask + online softmax
    const float scale = 0.08838834764831845f;  // 1/sqrt(128)
    float mnew[8];
#pragma unroll
    for (int r = 0; r < 8; ++r) {
      int q = qbase + r + 8 * hl;
      float v0 = st[0][r] * scale;
      float v1 = st[1][r] * scale;
      if (kt * 32 + ln > q)      v0 = -3.0e38f;
      if (kt * 32 + 16 + ln > q) v1 = -3.0e38f;
      st[0][r] = v0; st[1][r] = v1;
      float rm = fmaxf(v0, v1);
#pragma unroll
      for (int m = 1; m < 16; m <<= 1) rm = fmaxf(rm, __shfl_xor(rm, m, 32));
      mnew[r] = fmaxf(mrow[r], rm);
    }
#pragma unroll
    for (int r = 0; r < 8; ++r) {
      float a = __expf(mrow[r] - mnew[r]);
      float p0 = __expf(st[0][r] - mnew[r]);
      float p1 = __expf(st[1][r] - mnew[r]);
      st[0][r] = p0; st[1][r] = p1;
      float rsum = p0 + p1;
#pragma unroll
      for (int m = 1; m < 16; m <<= 1) rsum += __shfl_xor(rsum, m, 32);
      lrow[r] = lrow[r] * a + rsum;
      mrow[r] = mnew[r];
#pragma unroll
      for (int hf = 0; hf < 8; ++hf) o[hf][r] *= a;
    }

    // write P (per-wave private LDS region), then O += P V
#pragma unroll
    for (int r = 0; r < 8; ++r) {
      int m = r + 8 * hl;
      sP[w * 512 + m * 32 + ln]      = f2bf(st[0][r]);
      sP[w * 512 + m * 32 + 16 + ln] = f2bf(st[1][r]);
    }
    __syncthreads();
    v16bf pf = frag_a32(&sP[w * 512 + ln * 32]);
#pragma unroll
    for (int hf = 0; hf < 8; ++hf) {
      v16bf vfg = frag_b32_strided(&sV[hf * 16 + ln], 128);
      o[hf] = wmma_bf16(pf, vfg, o[hf]);
    }
    __syncthreads();
  }

  // epilogue: normalize and store bf16 [T, H*HD]
#pragma unroll
  for (int hf = 0; hf < 8; ++hf)
#pragma unroll
    for (int r = 0; r < 8; ++r) {
      int m = r + 8 * hl;
      float val = o[hf][r] / lrow[r];
      Ob[((size_t)b * kS + qbase + m) * (kH * kHD) + h * kHD + hf * 16 + ln] = f2bf(val);
    }
}

// ---------- router: logits, softmax, top-2 ----------
__global__ __launch_bounds__(256) void router_kernel(
    const bf16* __restrict__ X, const float* __restrict__ Wr,
    int* __restrict__ top_idx, float* __restrict__ top_w, int* __restrict__ counts) {
  int w = threadIdx.x >> 5, lane = threadIdx.x & 31;
  int t = blockIdx.x * 8 + w;
  float acc[kE];
#pragma unroll
  for (int e = 0; e < kE; ++e) acc[e] = 0.f;
  const bf16* x = X + (size_t)t * kD;
  for (int d = lane; d < kD; d += 32) {
    float xv = b2f(x[d]);
    const float* wr = Wr + (size_t)d * kE;
#pragma unroll
    for (int e = 0; e < kE; ++e) acc[e] += xv * wr[e];
  }
#pragma unroll
  for (int e = 0; e < kE; ++e)
#pragma unroll
    for (int m = 1; m < 32; m <<= 1) acc[e] += __shfl_xor(acc[e], m, 32);
  if (lane == 0) {
    float mx = acc[0];
#pragma unroll
    for (int e = 1; e < kE; ++e) mx = fmaxf(mx, acc[e]);
    float p[kE], s = 0.f;
#pragma unroll
    for (int e = 0; e < kE; ++e) { p[e] = __expf(acc[e] - mx); s += p[e]; }
#pragma unroll
    for (int e = 0; e < kE; ++e) p[e] /= s;
    int i0 = 0;
#pragma unroll
    for (int e = 1; e < kE; ++e) if (p[e] > p[i0]) i0 = e;
    int i1 = (i0 == 0) ? 1 : 0;
#pragma unroll
    for (int e = 0; e < kE; ++e) if (e != i0 && p[e] > p[i1]) i1 = e;
    float s2 = p[i0] + p[i1];
    top_idx[2 * t] = i0; top_idx[2 * t + 1] = i1;
    top_w[2 * t] = p[i0] / s2; top_w[2 * t + 1] = p[i1] / s2;
    atomicAdd(&counts[i0], 1);
    atomicAdd(&counts[i1], 1);
  }
}

__global__ void scan_kernel(const int* __restrict__ counts, int* __restrict__ offs) {
  if (threadIdx.x == 0) {
    int s = 0;
#pragma unroll
    for (int e = 0; e < kE; ++e) { offs[e] = s; s += counts[e]; }
    offs[kE] = s;
  }
}

// deterministic expert-major permutation via wave ballot prefix (1 block, wave e -> expert e)
__global__ __launch_bounds__(256) void scatter_kernel(
    const int* __restrict__ top_idx, const float* __restrict__ top_w,
    const int* __restrict__ offs, int* __restrict__ ptok, float* __restrict__ pw,
    int* __restrict__ pos) {
  int e = threadIdx.x >> 5;
  int lane = threadIdx.x & 31;
  int base = offs[e];
  int cnt = 0;
  for (int t0 = 0; t0 < kT; t0 += 32) {
    int t = t0 + lane;
    int slot = -1;
    if (top_idx[2 * t] == e) slot = 0;
    else if (top_idx[2 * t + 1] == e) slot = 1;
    unsigned long long mask = __ballot(slot >= 0);
    unsigned mlow = (unsigned)mask;
    int rank = __popc(mlow & ((1u << lane) - 1u));
    if (slot >= 0) {
      int p = base + cnt + rank;
      ptok[p] = t;
      pw[p] = top_w[2 * t + slot];
      pos[2 * t + slot] = p;
    }
    cnt += __popc(mlow);
  }
}

// ---------- MoE gate+up GEMM with fused SiLU ----------
__global__ __launch_bounds__(256) void moe_gate_up_kernel(
    const bf16* __restrict__ X, const float* __restrict__ Wg, const float* __restrict__ Wu,
    const int* __restrict__ ptok, const int* __restrict__ offs, bf16* __restrict__ Hbuf) {
  __shared__ __align__(32) bf16 sA[128 * 32];
  __shared__ __align__(32) bf16 sG[64 * 32];
  __shared__ __align__(32) bf16 sU[64 * 32];
  __shared__ int sTok[128];
  int e = blockIdx.z;
  int off = offs[e], cnt = offs[e + 1] - off;
  int mBase = blockIdx.y * 128;
  if (mBase >= cnt) return;
  int nBase = blockIdx.x * 64;
  const float* wg = Wg + (size_t)e * kD * kF;
  const float* wu = Wu + (size_t)e * kD * kF;
  if (threadIdx.x < 128) {
    int r = mBase + threadIdx.x;
    sTok[threadIdx.x] = (r < cnt) ? ptok[off + r] : -1;
  }
  __syncthreads();
  int w = threadIdx.x >> 5, lane = threadIdx.x & 31;
  int ln = lane & 15, hl = lane >> 4;
  v8f zero = {};
  v8f ag[4], au[4];
#pragma unroll
  for (int i = 0; i < 4; ++i) { ag[i] = zero; au[i] = zero; }

  for (int k0 = 0; k0 < kD; k0 += 32) {
    {  // gathered A tile (async DMA when available)
      int row = threadIdx.x >> 1;
      int ch = (threadIdx.x & 1) * 16;
      int tok = sTok[row];
      bf16* dst = &sA[row * 32 + ch];
      if (tok >= 0) {
        lds_copy16(dst, X + (size_t)tok * kD + k0 + ch);
      } else {
        v16bf zz = {};
        *(v16bf*)dst = zz;
      }
    }
    {  // gate + up weight tiles, f32 -> bf16, N-major
      int base = threadIdx.x * 8;
      int k = base >> 6;
      int n = base & 63;
      const float* g = wg + (size_t)(k0 + k) * kF + nBase + n;
      const float* u = wu + (size_t)(k0 + k) * kF + nBase + n;
#pragma unroll
      for (int i = 0; i < 8; ++i) {
        sG[(n + i) * 32 + k] = f2bf(g[i]);
        sU[(n + i) * 32 + k] = f2bf(u[i]);
      }
    }
    async_fence();
    __syncthreads();
    v16bf af = frag_a32(&sA[(w * 16 + ln) * 32]);
#pragma unroll
    for (int nS = 0; nS < 4; ++nS) {
      v16bf bg = frag_b32(&sG[(nS * 16 + ln) * 32]);
      ag[nS] = wmma_bf16(af, bg, ag[nS]);
      v16bf bu = frag_b32(&sU[(nS * 16 + ln) * 32]);
      au[nS] = wmma_bf16(af, bu, au[nS]);
    }
    __syncthreads();
  }
#pragma unroll
  for (int nS = 0; nS < 4; ++nS)
#pragma unroll
    for (int r = 0; r < 8; ++r) {
      int m = w * 16 + r + 8 * hl;
      if (mBase + m < cnt) {
        float g = ag[nS][r];
        float u = au[nS][r];
        float hv = (g / (1.0f + __expf(-g))) * u;   // silu(g) * u
        Hbuf[(size_t)(off + mBase + m) * kF + nBase + nS * 16 + ln] = f2bf(hv);
      }
    }
}

// ---------- MoE down GEMM, weighted, exclusive per-row output ----------
__global__ __launch_bounds__(256) void moe_down_kernel(
    const bf16* __restrict__ Hbuf, const float* __restrict__ Wd,
    const float* __restrict__ pw, const int* __restrict__ offs, float* __restrict__ Mo) {
  __shared__ __align__(32) bf16 sA[128 * 32];
  __shared__ __align__(32) bf16 sB[64 * 32];
  __shared__ float sW[128];
  int e = blockIdx.z;
  int off = offs[e], cnt = offs[e + 1] - off;
  int mBase = blockIdx.y * 128;
  if (mBase >= cnt) return;
  int nBase = blockIdx.x * 64;
  const float* wd = Wd + (size_t)e * kF * kD;
  if (threadIdx.x < 128) {
    int r = mBase + threadIdx.x;
    sW[threadIdx.x] = (r < cnt) ? pw[off + r] : 0.f;
  }
  __syncthreads();
  int w = threadIdx.x >> 5, lane = threadIdx.x & 31;
  int ln = lane & 15, hl = lane >> 4;
  v8f zero = {};
  v8f acc[4];
#pragma unroll
  for (int i = 0; i < 4; ++i) acc[i] = zero;

  for (int k0 = 0; k0 < kF; k0 += 32) {
    {
      int row = threadIdx.x >> 1;
      int ch = (threadIdx.x & 1) * 16;
      bf16* dst = &sA[row * 32 + ch];
      if (mBase + row < cnt) {
        lds_copy16(dst, Hbuf + (size_t)(off + mBase + row) * kF + k0 + ch);
      } else {
        v16bf zz = {};
        *(v16bf*)dst = zz;
      }
    }
    {
      int base = threadIdx.x * 8;
      int k = base >> 6;
      int n = base & 63;
      const float* src = wd + (size_t)(k0 + k) * kD + nBase + n;
#pragma unroll
      for (int i = 0; i < 8; ++i) sB[(n + i) * 32 + k] = f2bf(src[i]);
    }
    async_fence();
    __syncthreads();
    v16bf af = frag_a32(&sA[(w * 16 + ln) * 32]);
#pragma unroll
    for (int nS = 0; nS < 4; ++nS) {
      v16bf bfg = frag_b32(&sB[(nS * 16 + ln) * 32]);
      acc[nS] = wmma_bf16(af, bfg, acc[nS]);
    }
    __syncthreads();
  }
#pragma unroll
  for (int nS = 0; nS < 4; ++nS)
#pragma unroll
    for (int r = 0; r < 8; ++r) {
      int m = w * 16 + r + 8 * hl;
      if (mBase + m < cnt) {
        Mo[(size_t)(off + mBase + m) * kD + nBase + nS * 16 + ln] = acc[nS][r] * sW[m];
      }
    }
}

// ---------- final combine: out += moe(slot0) + moe(slot1) (deterministic) ----------
__global__ void moe_combine_kernel(const float* __restrict__ Mo, const int* __restrict__ pos,
                                   float* __restrict__ Out) {
  size_t i = (size_t)blockIdx.x * blockDim.x + threadIdx.x;
  if (i >= (size_t)kT * kD) return;
  size_t t = i / kD;
  int d = (int)(i - t * kD);
  int p0 = pos[2 * t], p1 = pos[2 * t + 1];
  Out[i] += Mo[(size_t)p0 * kD + d] + Mo[(size_t)p1 * kD + d];
}

// ---------------------------------------------------------------------------
extern "C" void kernel_launch(void* const* d_in, const int* in_sizes, int n_in,
                              void* d_out, int out_size, void* d_ws, size_t ws_size,
                              hipStream_t stream) {
  (void)in_sizes; (void)n_in; (void)out_size; (void)ws_size;
  const float* hidden   = (const float*)d_in[0];
  // d_in[1] = position_ids (always tiled arange(S); computed in-kernel)
  const float* ln1_w    = (const float*)d_in[2];
  const float* wq       = (const float*)d_in[3];
  const float* wk       = (const float*)d_in[4];
  const float* wv       = (const float*)d_in[5];
  const float* wo       = (const float*)d_in[6];
  const float* ln2_w    = (const float*)d_in[7];
  const float* router_w = (const float*)d_in[8];
  const float* w_gate   = (const float*)d_in[9];
  const float* w_up     = (const float*)d_in[10];
  const float* w_down   = (const float*)d_in[11];
  float* out = (float*)d_out;

  char* ws = (char*)d_ws;
  const size_t MB = 1024ull * 1024ull;
  bf16*  lnb  = (bf16*)(ws + 0);          // 16MB: ln1 output, later reused for ln2 output
  float* qf   = (float*)(ws + 16 * MB);   // 32MB: q f32 (later aliased by Hbuf)
  float* kf   = (float*)(ws + 48 * MB);   // 8MB
  float* vf   = (float*)(ws + 56 * MB);   // 8MB
  bf16*  qb   = (bf16*)(ws + 64 * MB);    // 16MB
  bf16*  kb   = (bf16*)(ws + 80 * MB);    // 4MB
  bf16*  vb   = (bf16*)(ws + 84 * MB);    // 4MB
  bf16*  ab   = (bf16*)(ws + 88 * MB);    // 16MB: attention output (bf16)
  bf16*  hbuf = (bf16*)(ws + 16 * MB);    // 32MB alias of qf (free after rope)
  float* mo   = (float*)(ws + 48 * MB);   // 64MB alias of kf..ab (free after attention)
  char* meta = ws + 112 * MB;
  int*   top_idx = (int*)(meta);                     // kT*2 ints
  float* top_w   = (float*)(meta + 32768);           // kT*2 floats
  int*   counts  = (int*)(meta + 65536);             // 8 ints
  int*   offs    = (int*)(meta + 65600);             // 9 ints
  int*   ptok    = (int*)(meta + 65664);             // 2*kT ints
  float* pw      = (float*)(meta + 65664 + 32768);   // 2*kT floats
  int*   pos     = (int*)(meta + 65664 + 65536);     // 2*kT ints

  // 1) LN1 -> bf16
  layernorm_kernel<<<kT, 256, 0, stream>>>(hidden, ln1_w, lnb);
  // 2) Q/K/V projections (clip +-8), f32 out
  gemm_kernel<true, false><<<dim3(16, 32), 256, 0, stream>>>(lnb, wq, nullptr, qf, kT, 2048, kD);
  gemm_kernel<true, false><<<dim3(4, 32), 256, 0, stream>>>(lnb, wk, nullptr, kf, kT, 512, kD);
  gemm_kernel<true, false><<<dim3(4, 32), 256, 0, stream>>>(lnb, wv, nullptr, vf, kT, 512, kD);
  // 3) RoPE + bf16 convert
  {
    size_t nq = (size_t)kT * kH * 64;
    rope_kernel<<<(unsigned)((nq + 255) / 256), 256, 0, stream>>>(qf, qb, kH);
    size_t nk = (size_t)kT * kHKV * 64;
    rope_kernel<<<(unsigned)((nk + 255) / 256), 256, 0, stream>>>(kf, kb, kHKV);
    size_t nv = (size_t)kT * kHKV * kHD;
    f32_to_bf16_kernel<<<(unsigned)((nv + 255) / 256), 256, 0, stream>>>(vf, vb, nv);
  }
  // 4) causal flash attention (WMMA QK^T and PV)
  flash_attn_kernel<<<dim3(kS / 128, kBATCH * kH), 256, 0, stream>>>(qb, kb, vb, ab);
  // 5) output projection + residual -> d_out (= h1)
  gemm_kernel<false, true><<<dim3(16, 32), 256, 0, stream>>>(ab, wo, hidden, out, kT, kD, 2048);
  // 6) LN2 -> bf16 (reuses lnb)
  layernorm_kernel<<<kT, 256, 0, stream>>>(out, ln2_w, lnb);
  // 7) router + deterministic expert-major permutation
  (void)hipMemsetAsync(counts, 0, 32, stream);
  router_kernel<<<kT / 8, 256, 0, stream>>>(lnb, router_w, top_idx, top_w, counts);
  scan_kernel<<<1, 32, 0, stream>>>(counts, offs);
  scatter_kernel<<<1, 256, 0, stream>>>(top_idx, top_w, offs, ptok, pw, pos);
  // 8) expert GEMMs
  moe_gate_up_kernel<<<dim3(kF / 64, 32, kE), 256, 0, stream>>>(lnb, w_gate, w_up, ptok, offs, hbuf);
  moe_down_kernel<<<dim3(kD / 64, 32, kE), 256, 0, stream>>>(hbuf, w_down, pw, offs, mo);
  // 9) combine: out = h1 + moe
  moe_combine_kernel<<<(unsigned)(((size_t)kT * kD + 255) / 256), 256, 0, stream>>>(mo, pos, out);
}
